// ChangepointDetector_16003048145387
// MI455X (gfx1250) — matrix operand order repaired and checked by reading, not compiled, not run on hardware
//
#include <hip/hip_runtime.h>
#include <hip/hip_bf16.h>
#include <cstdint>

// Problem dims (reference: B,S,I,W,H = 32,2048,32,20,128)
#define B_   32
#define S_   2048
#define I_   32
#define W_   20
#define H_   128
#define NW_  2029          // S - W + 1 sliding windows
#define T_   2008          // S - 2W valid changepoint positions
#define NTILE_ENC 127      // ceil(NW_/16)
#define NTILE_CMP 126      // ceil(T_/16)
#define LN_EPS_ 1e-5f

typedef __attribute__((ext_vector_type(16))) _Float16 v16h;
typedef __attribute__((ext_vector_type(8)))  _Float16 v8h;
typedef __attribute__((ext_vector_type(8)))  float    v8f;
typedef __attribute__((ext_vector_type(4)))  unsigned u32x4;
typedef __attribute__((ext_vector_type(8)))  int      i32x8;
typedef __attribute__((ext_vector_type(4)))  int      i32x4;

#if __has_builtin(__builtin_amdgcn_tensor_load_to_lds) && __has_builtin(__builtin_amdgcn_s_wait_tensorcnt)
#define USE_TDM 1
#else
#define USE_TDM 0
#endif

__device__ __forceinline__ float gelu_exact(float x) {
  return 0.5f * x * (1.0f + erff(x * 0.70710678118654752f));
}

// Build a 16-bit A/B fragment from two contiguous 16-byte chunks.
__device__ __forceinline__ v16h load_frag_pair(const _Float16* p0, const _Float16* p1) {
  v8h lo = *(const v8h*)p0;
  v8h hv = *(const v8h*)p1;
  return __builtin_shufflevector(lo, hv, 0,1,2,3,4,5,6,7,8,9,10,11,12,13,14,15);
}

__device__ __forceinline__ v8f wmma_f16(v16h a, v16h b, v8f c) {
  // D(f32 16x16) = A(f16 16x32) * B(f16 32x16) + C
  return __builtin_amdgcn_wmma_f32_16x16x32_f16(false, a, false, b, (short)0, c, false, false);
}

// ---------------------------------------------------------------- prep kernels

__global__ void zero_kernel(float* p, int n) {
  int i = blockIdx.x * blockDim.x + threadIdx.x;
  if (i < n) p[i] = 0.0f;
}

__global__ void cvt_f16_kernel(const float* __restrict__ src, _Float16* __restrict__ dst, int n) {
  int i = blockIdx.x * blockDim.x + threadIdx.x;
  if (i < n) dst[i] = (_Float16)src[i];
}

// Pack a (KC*32) x N fp32 weight into f16 WMMA B-fragment order:
// dst[((kc*F + f)*32 + lane)*16 + e] = W[(kc*32 + koff2 + e + 8*(e>=8))*N + f*16 + (lane&15)]
// with koff2 = 8*(lane>=16).  F = N/16 column tiles.
__global__ void pack_kernel(const float* __restrict__ w, _Float16* __restrict__ dst,
                            int KC, int F, int N) {
  int total = KC * F * 512;
  int i = blockIdx.x * blockDim.x + threadIdx.x;
  if (i >= total) return;
  int e    = i & 15;
  int lane = (i >> 4) & 31;
  int rest = i >> 9;
  int f    = rest % F;
  int kc   = rest / F;
  int v = e >> 1, hl = e & 1;
  int koff2 = (lane & 16) ? 8 : 0;
  int n = lane & 15;
  int k = kc * 32 + koff2 + 2 * v + hl + ((e >= 8) ? 8 : 0);
  dst[i] = (_Float16)w[(size_t)k * N + f * 16 + n];
}

// ---------------------------------------------------------------- encoder
// One wave per (batch, 16-window tile).  h = win @ W_enc + b; LN; GELU -> enc(f16)
// Packed W_enc K-slices are streamed into LDS by the Tensor Data Mover,
// double-buffered so slice kc+1 DMAs behind the WMMAs of slice kc.

__global__ __launch_bounds__(32)
void enc_kernel(const _Float16* __restrict__ xh, const _Float16* __restrict__ bpack,
                const float* __restrict__ b_enc, const float* __restrict__ gamma,
                const float* __restrict__ beta, _Float16* __restrict__ ench) {
  int wg = blockIdx.x;
  int b = wg / NTILE_ENC, tile = wg % NTILE_ENC;
  int t0 = tile * 16;
  int lane = threadIdx.x;
  int nl = lane & 15;
  int hig = lane >> 4;        // which 16-lane group
  int koff = hig * 8;         // A-fragment K offset per ISA layout

  v8f acc[8] = {};

#if USE_TDM
  __shared__ _Float16 bs[2][4096];        // double-buffered packed W_enc K-slice (2x8 KB)

  auto issue_tdm = [&](int kc) {
    unsigned long long ga = (unsigned long long)(uintptr_t)(bpack + (size_t)kc * 4096);
    unsigned lds_off = (unsigned)(uintptr_t)(&bs[kc & 1][0]);  // low 32 bits = LDS offset
    u32x4 g0;
    g0.x = 1u;                                      // count=1, user descriptor
    g0.y = lds_off;                                 // lds_addr
    g0.z = (unsigned)(ga & 0xFFFFFFFFull);          // global_addr[31:0]
    g0.w = (unsigned)((ga >> 32) & 0x1FFFFFFull) | (2u << 30);  // addr[56:32] | type=2
    i32x8 g1 = {0,0,0,0,0,0,0,0};
    g1[0] = (1 << 16);                              // data_size=1 (2 bytes/elem)
    g1[1] = (int)(4096u << 16);                     // tensor_dim0[15:0]
    g1[2] = (1 << 16);                              // tensor_dim0[31:16]=0 | tensor_dim1=1
    g1[3] = (int)(4096u << 16);                     // tile_dim0 = 4096
    g1[5] = 4096;                                   // tensor_dim0_stride
    i32x4 z4 = {0, 0, 0, 0};
    i32x8 z8 = {0,0,0,0,0,0,0,0};
    // Fence: no DS reads of the target buffer may still be in flight
    // (TDM LDS writes are unordered vs outstanding DS ops).
    asm volatile("s_wait_dscnt 0x0" ::: "memory");
#if defined(__clang_major__) && __clang_major__ >= 23
    __builtin_amdgcn_tensor_load_to_lds(g0, g1, z4, z4, z8, 0);
#else
    __builtin_amdgcn_tensor_load_to_lds(g0, g1, z4, z4, 0);
#endif
  };

  issue_tdm(0);                            // prologue: start slice 0
#endif

  for (int kc = 0; kc < W_; ++kc) {
    // A fragment: 16 window rows, K-chunk kc == row (t+kc) of x, 32 f16 features
    int r = t0 + nl + kc;
    if (r > S_ - 1) r = S_ - 1;           // clamp (rows past nwin are never stored)
    const _Float16* pa = xh + ((size_t)b * S_ + r) * I_ + koff;
    v16h af = load_frag_pair(pa, pa + 16);

#if USE_TDM
    if (kc + 1 < W_) {
      issue_tdm(kc + 1);                             // prefetch next slice (other buffer)
      __builtin_amdgcn_s_wait_tensorcnt(1);          // in-order: slice kc is resident
    } else {
      __builtin_amdgcn_s_wait_tensorcnt(0);          // last slice: drain
    }
    asm volatile("" ::: "memory");
    const _Float16* bb = &bs[kc & 1][0];
#else
    const _Float16* bb = bpack + (size_t)kc * 4096;
    if (kc + 1 < W_) __builtin_prefetch(bpack + (size_t)(kc + 1) * 4096, 0, 0);
#endif

#pragma unroll
    for (int f = 0; f < 8; ++f) {
      v16h bf = *(const v16h*)(bb + (f * 32 + lane) * 16);
      acc[f] = wmma_f16(af, bf, acc[f]);
    }
  }

  // Bias + LayerNorm + GELU.  C/D fragment: lane owns column n=f*16+nl,
  // rows m = v + 8*hig live in vector slot v.
  float bias[8], gm[8], bt[8];
#pragma unroll
  for (int f = 0; f < 8; ++f) {
    bias[f] = b_enc[f * 16 + nl];
    gm[f]   = gamma[f * 16 + nl];
    bt[f]   = beta[f * 16 + nl];
  }
#pragma unroll
  for (int f = 0; f < 8; ++f)
#pragma unroll
    for (int v = 0; v < 8; ++v) acc[f][v] += bias[f];

  float mu[8], rs[8];
#pragma unroll
  for (int v = 0; v < 8; ++v) {
    float s = 0.0f, q = 0.0f;
#pragma unroll
    for (int f = 0; f < 8; ++f) { float h = acc[f][v]; s += h; q += h * h; }
    // reduce across the 16 lanes that hold this row's columns
#pragma unroll
    for (int m = 1; m < 16; m <<= 1) {
      s += __shfl_xor(s, m, 32);
      q += __shfl_xor(q, m, 32);
    }
    float m1 = s * (1.0f / (float)H_);
    float var = q * (1.0f / (float)H_) - m1 * m1;
    mu[v] = m1;
    rs[v] = rsqrtf(var + LN_EPS_);
  }

#pragma unroll
  for (int f = 0; f < 8; ++f) {
#pragma unroll
    for (int v = 0; v < 8; ++v) {
      int t = t0 + v + 8 * hig;
      if (t < NW_) {
        float y = (acc[f][v] - mu[v]) * rs[v] * gm[f] + bt[f];
        ench[((size_t)b * NW_ + t) * H_ + f * 16 + nl] = (_Float16)gelu_exact(y);
      }
    }
  }
}

// ---------------------------------------------------------------- comparator
// One wave per (batch, 16-position tile):
// z1 = gelu([enc(t)||enc(t+W)] @ W1 + b1); z2 = gelu(z1 @ W2 + b2); p = sigmoid(z2@W3+b3)

__global__ __launch_bounds__(32)
void cmp_kernel(const _Float16* __restrict__ ench, const _Float16* __restrict__ bp1,
                const _Float16* __restrict__ bp2, const float* __restrict__ b1,
                const float* __restrict__ b2, const float* __restrict__ w3,
                const float* __restrict__ b3, float* __restrict__ out) {
  int wg = blockIdx.x;
  int b = wg / NTILE_CMP, tile = wg % NTILE_CMP;
  int t0 = tile * 16;
  int lane = threadIdx.x;
  int nl = lane & 15;
  int hig = lane >> 4;
  int koff = hig * 8;

  __shared__ _Float16 z1h[16 * H_];   // 4 KB: z1 tile in [row][col] f16 (for lane transpose)
  __shared__ float    z2f[16 * 64];   // 4 KB: z2 tile fp32

  // GEMM1: A = combined 16x256 (K chunks 0..3 from enc(t), 4..7 from enc(t+W))
  v8f acc1[8] = {};
#pragma unroll
  for (int kc = 0; kc < 8; ++kc) {
    int r = t0 + nl + ((kc >= 4) ? W_ : 0);
    if (r > NW_ - 1) r = NW_ - 1;
    const _Float16* pa = ench + ((size_t)b * NW_ + r) * H_ + (kc & 3) * 32 + koff;
    v16h af = load_frag_pair(pa, pa + 16);
#pragma unroll
    for (int f = 0; f < 8; ++f) {
      v16h bf = *(const v16h*)(bp1 + ((kc * 8 + f) * 32 + lane) * 16);
      acc1[f] = wmma_f16(af, bf, acc1[f]);
    }
  }

  // bias + GELU -> z1 (LDS, [m][c] layout).  Same-wave DS ops are in-order: no barrier.
#pragma unroll
  for (int f = 0; f < 8; ++f) {
    float bb = b1[f * 16 + nl];
#pragma unroll
    for (int v = 0; v < 8; ++v) {
      int m = v + 8 * hig;
      z1h[m * H_ + f * 16 + nl] = (_Float16)gelu_exact(acc1[f][v] + bb);
    }
  }

  // GEMM2: z1(16x128) @ W2(128x64)
  v8f acc2[4] = {};
#pragma unroll
  for (int kc = 0; kc < 4; ++kc) {
    const _Float16* pa = &z1h[nl * H_ + kc * 32 + koff];
    v16h af = load_frag_pair(pa, pa + 16);
#pragma unroll
    for (int f2 = 0; f2 < 4; ++f2) {
      v16h bf = *(const v16h*)(bp2 + ((kc * 4 + f2) * 32 + lane) * 16);
      acc2[f2] = wmma_f16(af, bf, acc2[f2]);
    }
  }

#pragma unroll
  for (int f2 = 0; f2 < 4; ++f2) {
    float bb = b2[f2 * 16 + nl];
#pragma unroll
    for (int v = 0; v < 8; ++v) {
      int m = v + 8 * hig;
      z2f[m * 64 + f2 * 16 + nl] = gelu_exact(acc2[f2][v] + bb);
    }
  }

  // Final 64-wide dot + sigmoid; lanes 0..15 each own one row.
  if (lane < 16) {
    int t = t0 + lane;
    if (t < T_) {
      float dot = b3[0];
#pragma unroll
      for (int c = 0; c < 64; ++c) dot += z2f[lane * 64 + c] * w3[c];
      float p = 1.0f / (1.0f + expf(-dot));
      size_t idx = (size_t)b * S_ + W_ + t;
      out[idx] = p;                                    // probs
      out[(size_t)B_ * S_ + idx] = (p > 0.5f) ? 1.0f : 0.0f;  // mask
    }
  }
}

// ---------------------------------------------------------------- launch

extern "C" void kernel_launch(void* const* d_in, const int* in_sizes, int n_in,
                              void* d_out, int out_size, void* d_ws, size_t ws_size,
                              hipStream_t stream) {
  (void)in_sizes; (void)n_in; (void)out_size; (void)ws_size;

  const float* x     = (const float*)d_in[0];
  const float* W_enc = (const float*)d_in[1];
  const float* b_enc = (const float*)d_in[2];
  const float* gamma = (const float*)d_in[3];
  const float* beta  = (const float*)d_in[4];
  const float* W1    = (const float*)d_in[5];
  const float* b1    = (const float*)d_in[6];
  const float* W2    = (const float*)d_in[7];
  const float* b2    = (const float*)d_in[8];
  const float* W3    = (const float*)d_in[9];
  const float* b3    = (const float*)d_in[10];
  float* out = (float*)d_out;

  char* ws = (char*)d_ws;
  size_t off = 0;
  auto alloc = [&](size_t bytes) -> void* {
    void* p = ws + off;
    off = (off + bytes + 255) & ~(size_t)255;
    return p;
  };
  _Float16* xh   = (_Float16*)alloc((size_t)B_ * S_ * I_ * 2);   // x as f16
  _Float16* bpE  = (_Float16*)alloc((size_t)640 * 128 * 2);      // packed W_enc
  _Float16* bp1  = (_Float16*)alloc((size_t)256 * 128 * 2);      // packed W1
  _Float16* bp2  = (_Float16*)alloc((size_t)128 * 64 * 2);       // packed W2
  _Float16* ench = (_Float16*)alloc((size_t)B_ * NW_ * H_ * 2);  // encoder output f16

  int nz = 2 * B_ * S_;
  zero_kernel<<<(nz + 255) / 256, 256, 0, stream>>>(out, nz);

  int nx = B_ * S_ * I_;
  cvt_f16_kernel<<<(nx + 255) / 256, 256, 0, stream>>>(x, xh, nx);

  pack_kernel<<<(20 * 8 * 512 + 255) / 256, 256, 0, stream>>>(W_enc, bpE, 20, 8, 128);
  pack_kernel<<<(8 * 8 * 512 + 255) / 256, 256, 0, stream>>>(W1, bp1, 8, 8, 128);
  pack_kernel<<<(4 * 4 * 512 + 255) / 256, 256, 0, stream>>>(W2, bp2, 4, 4, 64);

  enc_kernel<<<B_ * NTILE_ENC, 32, 0, stream>>>(xh, bpE, b_enc, gamma, beta, ench);
  cmp_kernel<<<B_ * NTILE_CMP, 32, 0, stream>>>(ench, bp1, bp2, b1, b2, W3, b3, out);
}
